// MultiAttentionHead_11149735100599
// MI455X (gfx1250) — matrix operand order repaired and checked by reading, not compiled, not run on hardware
//
#include <hip/hip_runtime.h>

// ---------------------------------------------------------------------------
// MultiHeadAttention for MI455X (gfx1250): bf16 WMMA + fp32 accumulate,
// flash-attention (online softmax), LDS staging of shared operands via
// gfx1250 async global->LDS copies (ASYNCcnt) with double buffering.
//
// Shapes: B=2, T=2048, E=1024, H=16, D=64.
// Workspace layout (48 MB total):
//   [ 0MB) xb     bf16 [4096][1024]        (x converted)
//   [ 8MB) wqT    bf16 [16][64][1024]      (Wq^T per head)
//   [10MB) wkT    bf16 [16][64][1024]
//   [12MB) wvT    bf16 [16][64][1024]
//   [14MB) woT    bf16 [1024][1024]        (Wo^T)
//   [16MB) qb     bf16 [2][16][2048][64]
//   [24MB) kb     bf16 [2][16][2048][64]
//   [32MB) vT     bf16 [2][16][64][2048]   (V^T for P@V B-fragments)
//   [40MB) comb   bf16 [2][2048][1024]     (concat heads)
// ---------------------------------------------------------------------------

typedef __attribute__((ext_vector_type(16))) __bf16 v16bf;
typedef __attribute__((ext_vector_type(8)))  __bf16 v8bf;
typedef __attribute__((ext_vector_type(8)))  float  v8f;
typedef __attribute__((ext_vector_type(4)))  int    v4i;

#define WMMA_BF16(a, b, c) \
    __builtin_amdgcn_wmma_f32_16x16x32_bf16(false, (a), false, (b), (short)0, (c), false, false)

// ---- async global->LDS (gfx1250). Fallback: synchronous copy via VGPRs. ----
#if defined(__has_builtin)
#  if __has_builtin(__builtin_amdgcn_global_load_async_to_lds_b128) && \
      __has_builtin(__builtin_amdgcn_s_wait_asynccnt)
#    define HAVE_ASYNC_LDS 1
#  endif
#endif
#ifndef HAVE_ASYNC_LDS
#  define HAVE_ASYNC_LDS 0
#endif

__device__ __forceinline__ void async16(const void* g, void* l) {
#if HAVE_ASYNC_LDS
    __builtin_amdgcn_global_load_async_to_lds_b128(
        (__attribute__((address_space(1))) v4i*)g,
        (__attribute__((address_space(3))) v4i*)l, 0, 0);
#else
    *(v8bf*)l = *(const v8bf*)g;
#endif
}
__device__ __forceinline__ void wait_async() {
#if HAVE_ASYNC_LDS
    __builtin_amdgcn_s_wait_asynccnt(0);
#endif
}

// Copy a 64-row x 128-byte tile (8KB) global->LDS, 256 threads, 32B/thread.
__device__ __forceinline__ void copy_tile_64x128B(const __bf16* g, size_t gpitch_bytes,
                                                  __bf16* l, int tid) {
    const int c0  = tid * 32;
    const int row = c0 >> 7, col = c0 & 127;
    const char* gp = (const char*)g + (size_t)row * gpitch_bytes + col;
    char* lp = (char*)l + c0;
    async16(gp, lp);
    async16(gp + 16, lp + 16);
}

// ---- WMMA fragment loaders (wave32) ----
// A-matrix 16x32 (row-major, stride ld): lanes 0-15 row=lane K=0..15;
// lanes 16-31 row=lane-16 K=16..31.
__device__ __forceinline__ v16bf load_a16x32(const __bf16* base, int ld, int lane) {
    const int r   = lane & 15;
    const int sel = (lane >> 4) << 4;
    const __bf16* p = base + r * ld;
    v8bf lo = *(const v8bf*)(p + sel);
    v8bf hi = *(const v8bf*)(p + sel + 8);
    return __builtin_shufflevector(lo, hi, 0,1,2,3,4,5,6,7,8,9,10,11,12,13,14,15);
}
// B-matrix 32x16 from B^T (row-major [N][K], stride ld): lanes 0-15 col=lane
// K=0..15; lanes 16-31 col=lane-16 K=16..31.  One contiguous 32B per lane.
__device__ __forceinline__ v16bf load_bT32x16(const __bf16* baseT, int ld, int lane) {
    const int n    = lane & 15;
    const int koff = (lane >> 4) << 4;
    const __bf16* p = baseT + n * ld + koff;
    v8bf lo = *(const v8bf*)(p);
    v8bf hi = *(const v8bf*)(p + 8);
    return __builtin_shufflevector(lo, hi, 0,1,2,3,4,5,6,7,8,9,10,11,12,13,14,15);
}

// ---------------------------------------------------------------------- cvt
__global__ __launch_bounds__(256) void cvt_f32_bf16(const float* __restrict__ in,
                                                    __bf16* __restrict__ out, int n) {
    int i = blockIdx.x * 256 + threadIdx.x;
    if (i < n) out[i] = (__bf16)in[i];
}

// in: [batch][rows][cols] fp32  ->  out: [batch][cols][rows] bf16
__global__ __launch_bounds__(256) void cvt_transpose(const float* __restrict__ in,
                                                     __bf16* __restrict__ out,
                                                     int rows, int cols) {
    int i = blockIdx.x * 256 + threadIdx.x;
    int bsz = rows * cols;
    if (i >= bsz) return;
    in  += (size_t)blockIdx.y * bsz;
    out += (size_t)blockIdx.y * bsz;
    int r = i / cols, c = i % cols;
    out[(size_t)c * rows + r] = (__bf16)in[i];
}

// ----------------------------------------------------------------- QKV GEMM
// grid (32, 16, 3): 128-row M-tile, head, {q,k,v}. Weights (128KB) staged in
// LDS via async copy; A-fragment software-pipelined one K-step ahead.
__global__ __launch_bounds__(256) void qkv_proj_kernel(
    const __bf16* __restrict__ xb,
    const __bf16* __restrict__ wqT, const __bf16* __restrict__ wkT,
    const __bf16* __restrict__ wvT,
    __bf16* __restrict__ qb, __bf16* __restrict__ kb, __bf16* __restrict__ vT) {
    __shared__ __align__(16) __bf16 ldsw[64 * 1024];   // 128 KB weight slice
    const int tid  = threadIdx.x;
    const int lane = tid & 31;
    const int wave = tid >> 5;
    const int row0 = blockIdx.x * 128 + wave * 16;     // 0..4095 (= b*2048 + t)
    const int h    = blockIdx.y;
    const int z    = blockIdx.z;                       // 0=q 1=k 2=v
    const __bf16* wT = (z == 0 ? wqT : (z == 1 ? wkT : wvT)) + h * (64 * 1024);

    for (int off = tid * 16; off < 64 * 1024 * 2; off += 256 * 16)
        async16((const char*)wT + off, (char*)ldsw + off);
    wait_async();
    __syncthreads();

    v8f c[4] = {};
    v16bf a = load_a16x32(xb + (size_t)row0 * 1024, 1024, lane);
    for (int kc = 0; kc < 1024; kc += 32) {
        v16bf an = a;
        if (kc + 32 < 1024)
            an = load_a16x32(xb + (size_t)row0 * 1024 + kc + 32, 1024, lane);
#pragma unroll
        for (int f = 0; f < 4; ++f) {
            v16bf b = load_bT32x16(ldsw + (f * 16) * 1024 + kc, 1024, lane);
            c[f] = WMMA_BF16(a, b, c[f]);
        }
        a = an;
    }
    const int n = lane & 15, half = lane >> 4;
    const int bidx = row0 >> 11, t0 = row0 & 2047;
    if (z < 2) {  // q,k -> [b][h][t][d]
        __bf16* op = (z == 0 ? qb : kb) + ((size_t)(bidx * 16 + h) * 2048 + t0) * 64;
#pragma unroll
        for (int f = 0; f < 4; ++f)
#pragma unroll
            for (int r = 0; r < 8; ++r)
                op[(r + 8 * half) * 64 + f * 16 + n] = (__bf16)c[f][r];
    } else {      // v -> transposed [b][h][d][t]
        __bf16* op = vT + (size_t)(bidx * 16 + h) * (64 * 2048);
#pragma unroll
        for (int f = 0; f < 4; ++f)
#pragma unroll
            for (int r = 0; r < 8; ++r)
                op[(size_t)(f * 16 + n) * 2048 + t0 + 8 * half + r] = (__bf16)c[f][r];
    }
}

// ------------------------------------------------------------ flash attention
// grid (16, 16, 2). K/V^T tiles (64x64) double-buffered in LDS via async
// copies overlapped with WMMA+softmax. Wave owns 16 q-rows x D=64.
__global__ __launch_bounds__(256) void attn_kernel(
    const __bf16* __restrict__ qb, const __bf16* __restrict__ kb,
    const __bf16* __restrict__ vT, __bf16* __restrict__ comb) {
    __shared__ __align__(16) __bf16 lds_k[2][64 * 64];   // 2 x 8 KB
    __shared__ __align__(16) __bf16 lds_v[2][64 * 64];   // 2 x 8 KB
    __shared__ __align__(16) __bf16 lds_p[8][16 * 64];   // 16 KB P scratch

    const int tid  = threadIdx.x;
    const int lane = tid & 31;
    const int wave = tid >> 5;
    const int b = blockIdx.z, h = blockIdx.y;
    const int q0 = blockIdx.x * 128 + wave * 16;
    const int bh = b * 16 + h;
    const __bf16* qh = qb + (size_t)bh * 2048 * 64;
    const __bf16* kh = kb + (size_t)bh * 2048 * 64;
    const __bf16* vh = vT + (size_t)bh * 64 * 2048;
    const int n = lane & 15, half = lane >> 4;

    v16bf aq0 = load_a16x32(qh + (size_t)q0 * 64 + 0,  64, lane);
    v16bf aq1 = load_a16x32(qh + (size_t)q0 * 64 + 32, 64, lane);

    v8f acc[4] = {};
    float rmax[8], rsum[8];
#pragma unroll
    for (int r = 0; r < 8; ++r) { rmax[r] = -__builtin_inff(); rsum[r] = 0.f; }
    const float scale = 0.125f;                        // 1/sqrt(64)

    // prime buffer 0
    copy_tile_64x128B(kh, 128, lds_k[0], tid);         // K rows contiguous
    copy_tile_64x128B(vh, 4096, lds_v[0], tid);        // V^T rows stride T*2

    for (int it = 0; it < 32; ++it) {
        const int kv0 = it * 64;
        const int buf = it & 1;
        wait_async();
        __syncthreads();                                // tile ready; prev buf free
        if (it + 1 < 32) {                              // prefetch next tile
            copy_tile_64x128B(kh + (size_t)(kv0 + 64) * 64, 128, lds_k[buf ^ 1], tid);
            copy_tile_64x128B(vh + (kv0 + 64),          4096, lds_v[buf ^ 1], tid);
        }
        const __bf16* kt = lds_k[buf];
        const __bf16* vt = lds_v[buf];

        // ---- S = scale * Q @ K^T  (16 x 64), 8 WMMAs, B from LDS
        v8f s[4] = {};
#pragma unroll
        for (int f = 0; f < 4; ++f) {
            v16bf bk0 = load_bT32x16(kt + (f * 16) * 64 + 0,  64, lane);
            s[f] = WMMA_BF16(aq0, bk0, s[f]);
            v16bf bk1 = load_bT32x16(kt + (f * 16) * 64 + 32, 64, lane);
            s[f] = WMMA_BF16(aq1, bk1, s[f]);
        }
#pragma unroll
        for (int f = 0; f < 4; ++f) s[f] *= scale;

        // ---- online softmax (rows split across lane halves)
#pragma unroll
        for (int r = 0; r < 8; ++r) {
            float m = fmaxf(fmaxf(s[0][r], s[1][r]), fmaxf(s[2][r], s[3][r]));
            m = fmaxf(m, __shfl_xor(m, 1, 32));
            m = fmaxf(m, __shfl_xor(m, 2, 32));
            m = fmaxf(m, __shfl_xor(m, 4, 32));
            m = fmaxf(m, __shfl_xor(m, 8, 32));
            float mnew = fmaxf(rmax[r], m);
            float corr = __expf(rmax[r] - mnew);
            rmax[r] = mnew;
            float ps = 0.f;
#pragma unroll
            for (int f = 0; f < 4; ++f) {
                float p = __expf(s[f][r] - mnew);
                s[f][r] = p;
                ps += p;
            }
            ps += __shfl_xor(ps, 1, 32);
            ps += __shfl_xor(ps, 2, 32);
            ps += __shfl_xor(ps, 4, 32);
            ps += __shfl_xor(ps, 8, 32);
            rsum[r] = rsum[r] * corr + ps;
#pragma unroll
            for (int f = 0; f < 4; ++f) acc[f][r] *= corr;
        }

        // ---- re-layout P (C-frag -> A-frag) via per-wave LDS (wave-local,
        // LDS ops are in-order within a wave; no barrier needed)
        __bf16* pl = lds_p[wave];
#pragma unroll
        for (int f = 0; f < 4; ++f)
#pragma unroll
            for (int r = 0; r < 8; ++r)
                pl[(r + 8 * half) * 64 + f * 16 + n] = (__bf16)s[f][r];

        // ---- acc += P @ V  (K = 64 kv cols), 8 WMMAs, A and B from LDS
#pragma unroll
        for (int kc = 0; kc < 64; kc += 32) {
            v16bf ap = load_a16x32(pl + kc, 64, lane);
#pragma unroll
            for (int f = 0; f < 4; ++f) {
                v16bf bv = load_bT32x16(vt + (f * 16) * 64 + kc, 64, lane);
                acc[f] = WMMA_BF16(ap, bv, acc[f]);
            }
        }
    }

    // ---- normalize and write concat-head output [b][t][h*64+d]
    __bf16* op = comb + ((size_t)b * 2048 + q0) * 1024 + h * 64;
#pragma unroll
    for (int r = 0; r < 8; ++r) {
        float inv = 1.0f / rsum[r];
#pragma unroll
        for (int f = 0; f < 4; ++f)
            op[(size_t)(r + 8 * half) * 1024 + f * 16 + n] = (__bf16)(acc[f][r] * inv);
    }
}

// --------------------------------------------------------- output projection
// grid (32, 16): out = comb @ Wo + x (fp32). Wo^T slice (128KB) staged in LDS.
__global__ __launch_bounds__(256) void oproj_kernel(
    const __bf16* __restrict__ comb, const __bf16* __restrict__ woT,
    const float* __restrict__ x, float* __restrict__ out) {
    __shared__ __align__(16) __bf16 ldsw[64 * 1024];
    const int tid  = threadIdx.x;
    const int lane = tid & 31;
    const int wave = tid >> 5;
    const int row0 = blockIdx.x * 128 + wave * 16;
    const int n0   = blockIdx.y * 64;

    const __bf16* wslice = woT + (size_t)n0 * 1024;    // rows n0..n0+63, contiguous
    for (int off = tid * 16; off < 64 * 1024 * 2; off += 256 * 16)
        async16((const char*)wslice + off, (char*)ldsw + off);
    wait_async();
    __syncthreads();

    v8f c[4] = {};
    v16bf a = load_a16x32(comb + (size_t)row0 * 1024, 1024, lane);
    for (int kc = 0; kc < 1024; kc += 32) {
        v16bf an = a;
        if (kc + 32 < 1024)
            an = load_a16x32(comb + (size_t)row0 * 1024 + kc + 32, 1024, lane);
#pragma unroll
        for (int f = 0; f < 4; ++f) {
            v16bf b = load_bT32x16(ldsw + (f * 16) * 1024 + kc, 1024, lane);
            c[f] = WMMA_BF16(a, b, c[f]);
        }
        a = an;
    }
    const int n = lane & 15, half = lane >> 4;
#pragma unroll
    for (int f = 0; f < 4; ++f)
#pragma unroll
        for (int r = 0; r < 8; ++r) {
            size_t idx = (size_t)(row0 + r + 8 * half) * 1024 + n0 + f * 16 + n;
            out[idx] = c[f][r] + x[idx];
        }
}

// ---------------------------------------------------------------------------
extern "C" void kernel_launch(void* const* d_in, const int* in_sizes, int n_in,
                              void* d_out, int out_size, void* d_ws, size_t ws_size,
                              hipStream_t stream) {
    const float* x  = (const float*)d_in[0];   // [2,2048,1024]
    const float* Wq = (const float*)d_in[1];   // [16,1024,64]
    const float* Wk = (const float*)d_in[2];
    const float* Wv = (const float*)d_in[3];
    const float* Wo = (const float*)d_in[4];   // [1024,1024]
    float* out = (float*)d_out;

    char* ws = (char*)d_ws;
    const size_t MB = 1ull << 20;
    __bf16* xb   = (__bf16*)(ws + 0 * MB);
    __bf16* wqT  = (__bf16*)(ws + 8 * MB);
    __bf16* wkT  = (__bf16*)(ws + 10 * MB);
    __bf16* wvT  = (__bf16*)(ws + 12 * MB);
    __bf16* woT  = (__bf16*)(ws + 14 * MB);
    __bf16* qb   = (__bf16*)(ws + 16 * MB);
    __bf16* kb   = (__bf16*)(ws + 24 * MB);
    __bf16* vT   = (__bf16*)(ws + 32 * MB);
    __bf16* comb = (__bf16*)(ws + 40 * MB);

    // 1) precision conversion / weight transposes
    cvt_f32_bf16<<<16384, 256, 0, stream>>>(x, xb, 2 * 2048 * 1024);
    cvt_transpose<<<dim3(256, 16), 256, 0, stream>>>(Wq, wqT, 1024, 64);
    cvt_transpose<<<dim3(256, 16), 256, 0, stream>>>(Wk, wkT, 1024, 64);
    cvt_transpose<<<dim3(256, 16), 256, 0, stream>>>(Wv, wvT, 1024, 64);
    cvt_transpose<<<dim3(4096, 1), 256, 0, stream>>>(Wo, woT, 1024, 1024);

    // 2) Q/K/V projections (WMMA, LDS-staged weights)
    qkv_proj_kernel<<<dim3(32, 16, 3), 256, 0, stream>>>(xb, wqT, wkT, wvT, qb, kb, vT);

    // 3) flash attention (WMMA + online softmax, async double-buffered K/V)
    attn_kernel<<<dim3(16, 16, 2), 256, 0, stream>>>(qb, kb, vT, comb);

    // 4) output projection + residual (WMMA, fp32 out)
    oproj_kernel<<<dim3(32, 16), 256, 0, stream>>>(comb, woT, x, out);
}
